// EGNNLayer_29008209118006
// MI455X (gfx1250) — matrix operand order repaired and checked by reading, not compiled, not run on hardware
//
#include <hip/hip_runtime.h>
#include <math.h>

typedef __attribute__((ext_vector_type(16))) _Float16 v16h;
typedef __attribute__((ext_vector_type(8)))  _Float16 v8h;
typedef __attribute__((ext_vector_type(8)))  float    v8f;

// ---------------------------------------------------------------------------
// WMMA helpers (CDNA5 wave32, 16x16x32 f16 -> f32)
// ---------------------------------------------------------------------------
__device__ __forceinline__ v8f wmma_f16(v16h a, v16h b, v8f c) {
  return __builtin_amdgcn_wmma_f32_16x16x32_f16(false, a, false, b, (short)0, c,
                                                false, false);
}

// A fragment (16x32, 16-bit). lane l holds row m = l%16;
// halves 0..7 = K off+0..7, halves 8..15 = K off+16..23 (off = 8 if lane>=16).
__device__ __forceinline__ v16h load_a_frag(const _Float16* base, int stride,
                                            int lane) {
  const _Float16* p = base + (lane & 15) * stride + ((lane >> 4) << 3);
  v8h lo = *(const v8h*)(p);
  v8h hi = *(const v8h*)(p + 16);
  v16h r;
#pragma unroll
  for (int i = 0; i < 8; ++i) { r[i] = lo[i]; r[i + 8] = hi[i]; }
  return r;
}

// B fragment (32x16, 16-bit): lane l = column n0 + l%16; halves 0..7 = K k0+off+0..7,
// halves 8..15 = K k0+off+16..23 (off = 8 if lane>=16); weights pre-transposed WT[N][K].
__device__ __forceinline__ v16h load_b_frag(const _Float16* __restrict__ wT,
                                            int K, int n0, int k0, int lane) {
  const _Float16* p =
      wT + (size_t)(n0 + (lane & 15)) * K + (k0 + ((lane >> 4) << 3));
  v8h lo = *(const v8h*)(p);
  v8h hi = *(const v8h*)(p + 16);
  v16h r;
#pragma unroll
  for (int i = 0; i < 8; ++i) { r[i] = lo[i]; r[i + 8] = hi[i]; }
  return r;
}

// Per-lane async DMA: 16 bytes global -> LDS, tracked by ASYNCcnt.
__device__ __forceinline__ void async_copy_b128(unsigned lds_off,
                                                const void* gaddr) {
  asm volatile("global_load_async_to_lds_b128 %0, %1, off"
               :
               : "v"(lds_off), "v"(gaddr)
               : "memory");
}
__device__ __forceinline__ void wait_async_zero() {
  asm volatile("s_wait_asynccnt 0x0" ::: "memory");
}
// Truncate a generic (flat) LDS pointer to its 32-bit LDS byte offset.
__device__ __forceinline__ unsigned lds_offset_of(const void* p) {
  return (unsigned)(uintptr_t)p;
}

// ---------------------------------------------------------------------------
// Prep kernels
// ---------------------------------------------------------------------------
__global__ void zero_f32(float* __restrict__ p, size_t n) {
  size_t i = (size_t)blockIdx.x * blockDim.x + threadIdx.x;
  if (i < n) p[i] = 0.0f;
}

__global__ void f32_to_f16(const float* __restrict__ in,
                           _Float16* __restrict__ out, size_t n) {
  size_t i = (size_t)blockIdx.x * blockDim.x + threadIdx.x;
  if (i < n) out[i] = (_Float16)in[i];
}

// in: [K][Nn] f32 row-major  ->  out: [Nn][K] f16 row-major (transposed)
__global__ void transpose_to_half(const float* __restrict__ in,
                                  _Float16* __restrict__ out, int K, int Nn) {
  int idx = blockIdx.x * blockDim.x + threadIdx.x;
  int total = K * Nn;
  if (idx < total) {
    int n = idx / K;
    int k = idx - n * K;
    out[idx] = (_Float16)in[(size_t)k * Nn + n];
  }
}

// ---------------------------------------------------------------------------
// Edge kernel: fused message MLP + segment-sum aggregation
//   128 edges/block, 256 threads = 8 waves; wave w owns edge rows 16w..16w+15
// ---------------------------------------------------------------------------
__global__ __launch_bounds__(256) void egnn_edge_kernel(
    const _Float16* __restrict__ xh, const float* __restrict__ pos,
    const int* __restrict__ eidx, int E,
    const float* __restrict__ msgW1,  // dist row: msgW1[128*256 + n]
    const float* __restrict__ b1, const float* __restrict__ b2,
    const _Float16* __restrict__ W1aT, const _Float16* __restrict__ W1bT,
    const _Float16* __restrict__ W2T, float* __restrict__ agg,
    float* __restrict__ posSum, float* __restrict__ cntv) {
  extern __shared__ char smem[];
  _Float16* sXi = (_Float16*)smem;           // [128][64]
  _Float16* sXj = sXi + 128 * 64;            // [128][64]
  _Float16* sH  = sXj + 128 * 64;            // [128][256]
  float* sDist  = (float*)(sH + 128 * 256);  // [128]
  float* sPosIJ = sDist + 128;               // [128][3]
  float* sRowSq = sPosIJ + 128 * 3;          // [128]
  int* sI = (int*)(sRowSq + 128);            // [128]
  int* sJ = sI + 128;                        // [128]

  const int t  = threadIdx.x;
  const int e0 = blockIdx.x * 128;

  // ---- stage edge meta (indices, pos_ij, dist) ----
  if (t < 128) {
    int e = e0 + t;
    int vi = 0, vj = 0;
    bool valid = (e < E);
    if (valid) { vi = eidx[e]; vj = eidx[E + e]; }
    sI[t] = valid ? vi : -1;
    sJ[t] = vj;
    float dx = pos[(size_t)vj * 3 + 0] - pos[(size_t)vi * 3 + 0];
    float dy = pos[(size_t)vj * 3 + 1] - pos[(size_t)vi * 3 + 1];
    float dz = pos[(size_t)vj * 3 + 2] - pos[(size_t)vi * 3 + 2];
    sPosIJ[t * 3 + 0] = dx;
    sPosIJ[t * 3 + 1] = dy;
    sPosIJ[t * 3 + 2] = dz;
    sDist[t] = sqrtf(dx * dx + dy * dy + dz * dz);
  }
  __syncthreads();

  // ---- gather x[ei], x[ej] rows (f16, 128 B each) via async DMA to LDS ----
  // 128 rows x 8 chunks of 16 B per matrix; 256 threads -> 4 chunks each.
  {
    unsigned ldsXi = lds_offset_of(sXi);
    unsigned ldsXj = lds_offset_of(sXj);
#pragma unroll
    for (int it = 0; it < 4; ++it) {
      int c = t + it * 256;
      int e = c >> 3, q = c & 7;
      int vi = sI[e] < 0 ? 0 : sI[e];
      int vj = sJ[e];
      const char* gi = (const char*)(xh + (size_t)vi * 64) + q * 16;
      const char* gj = (const char*)(xh + (size_t)vj * 64) + q * 16;
      unsigned d = (unsigned)(e * 128 + q * 16);
      async_copy_b128(ldsXi + d, gi);
      async_copy_b128(ldsXj + d, gj);
    }
    wait_async_zero();
  }
  __syncthreads();

  const int w = t >> 5, l = t & 31;
  const int lm = l & 15;
  const int rowoff = (l >> 4) << 3;  // 0 or 8

  // ---- phase 1: H = relu(Xi*W1a + Xj*W1b + dist (x) w1_last + b1) ----
  {
    v16h ai0 = load_a_frag(&sXi[(16 * w) * 64 + 0], 64, l);
    v16h ai1 = load_a_frag(&sXi[(16 * w) * 64 + 32], 64, l);
    v16h aj0 = load_a_frag(&sXj[(16 * w) * 64 + 0], 64, l);
    v16h aj1 = load_a_frag(&sXj[(16 * w) * 64 + 32], 64, l);
#pragma unroll
    for (int ct = 0; ct < 16; ++ct) {
      int n = ct * 16 + lm;
      float wl = msgW1[128 * 256 + n];
      float bb = b1[n];
      v8f c;
#pragma unroll
      for (int r = 0; r < 8; ++r)
        c[r] = sDist[16 * w + r + rowoff] * wl + bb;
      c = wmma_f16(ai0, load_b_frag(W1aT, 64, ct * 16, 0, l), c);
      c = wmma_f16(ai1, load_b_frag(W1aT, 64, ct * 16, 32, l), c);
      c = wmma_f16(aj0, load_b_frag(W1bT, 64, ct * 16, 0, l), c);
      c = wmma_f16(aj1, load_b_frag(W1bT, 64, ct * 16, 32, l), c);
#pragma unroll
      for (int r = 0; r < 8; ++r)
        sH[(16 * w + r + rowoff) * 256 + n] = (_Float16)fmaxf(c[r], 0.0f);
    }
  }
  // No barrier: wave w reads only its own H row-tile below.

  // ---- phase 2: MSG = H*W2 + b2, K=256 ----
  v8f acc[8];
#pragma unroll
  for (int ct = 0; ct < 8; ++ct) {
    float bb = b2[ct * 16 + lm];
#pragma unroll
    for (int r = 0; r < 8; ++r) acc[ct][r] = bb;
  }
#pragma unroll
  for (int kk = 0; kk < 8; ++kk) {
    v16h a = load_a_frag(&sH[(16 * w) * 256 + kk * 32], 256, l);
#pragma unroll
    for (int ct = 0; ct < 8; ++ct)
      acc[ct] = wmma_f16(a, load_b_frag(W2T, 256, ct * 16, kk * 32, l), acc[ct]);
  }

  // ---- epilogue: segment-sum atomics + row sum-of-squares ----
  float sq[8];
#pragma unroll
  for (int r = 0; r < 8; ++r) sq[r] = 0.0f;
#pragma unroll
  for (int ct = 0; ct < 8; ++ct) {
    int n = ct * 16 + lm;
#pragma unroll
    for (int r = 0; r < 8; ++r) {
      float v = acc[ct][r];
      sq[r] += v * v;
      int m = 16 * w + r + rowoff;
      int node = sI[m];
      if (node >= 0) unsafeAtomicAdd(&agg[(size_t)node * 128 + n], v);
    }
  }
#pragma unroll
  for (int r = 0; r < 8; ++r) {
#pragma unroll
    for (int off = 1; off < 16; off <<= 1)
      sq[r] += __shfl_xor(sq[r], off, 32);
  }
  if (lm == 0) {
#pragma unroll
    for (int r = 0; r < 8; ++r) sRowSq[16 * w + r + rowoff] = sq[r];
  }
  __syncthreads();

  if (t < 128 && sI[t] >= 0) {
    int node = sI[t];
    float pw = tanhf(sqrtf(sRowSq[t]));
    unsafeAtomicAdd(&posSum[(size_t)node * 3 + 0], pw * sPosIJ[t * 3 + 0]);
    unsafeAtomicAdd(&posSum[(size_t)node * 3 + 1], pw * sPosIJ[t * 3 + 1]);
    unsafeAtomicAdd(&posSum[(size_t)node * 3 + 2], pw * sPosIJ[t * 3 + 2]);
    unsafeAtomicAdd(&cntv[node], 1.0f);
  }
}

// ---------------------------------------------------------------------------
// Node kernel: x_proj, update MLP, ELU residual, pos update
// ---------------------------------------------------------------------------
__global__ __launch_bounds__(256) void egnn_node_kernel(
    const _Float16* __restrict__ xh, const float* __restrict__ pos,
    const float* __restrict__ agg, const float* __restrict__ posSum,
    const float* __restrict__ cntv, const float* __restrict__ ub1,
    const float* __restrict__ ub2, const _Float16* __restrict__ ProjT,
    const _Float16* __restrict__ U1T, const _Float16* __restrict__ U2T,
    float* __restrict__ xout, float* __restrict__ posout, int Nn) {
  extern __shared__ char smem[];
  _Float16* sX  = (_Float16*)smem;  // [128][64]
  _Float16* sU  = sX + 128 * 64;    // [128][256]: cols 0..127 xproj, 128..255 agg
  _Float16* sH2 = sU + 128 * 256;   // [128][256]

  const int t  = threadIdx.x;
  const int n0 = blockIdx.x * 128;

  // ---- stage x rows via async DMA (f16 source), agg via convert ----
  {
    unsigned ldsX = lds_offset_of(sX);
#pragma unroll
    for (int it = 0; it < 4; ++it) {
      int c = t + it * 256;
      int m = c >> 3, q = c & 7;
      int g = n0 + m;
      int gs = (g < Nn) ? g : 0;  // out-of-range rows compute garbage, writes guarded
      const char* gp = (const char*)(xh + (size_t)gs * 64) + q * 16;
      async_copy_b128(ldsX + (unsigned)(m * 128 + q * 16), gp);
    }
  }
  {
    int m = t >> 1, half = t & 1;
    int g = n0 + m;
    _Float16* du = &sU[m * 256 + 128 + half * 64];
    if (g < Nn) {
      const float4* sa = (const float4*)(agg + (size_t)g * 128 + half * 64);
#pragma unroll
      for (int q = 0; q < 16; ++q) {
        float4 v = sa[q];
        du[4 * q + 0] = (_Float16)v.x; du[4 * q + 1] = (_Float16)v.y;
        du[4 * q + 2] = (_Float16)v.z; du[4 * q + 3] = (_Float16)v.w;
      }
    } else {
#pragma unroll
      for (int q = 0; q < 64; ++q) du[q] = (_Float16)0.0f;
    }
  }
  wait_async_zero();
  __syncthreads();

  const int w = t >> 5, l = t & 31;
  const int lm = l & 15;
  const int rowoff = (l >> 4) << 3;

  // ---- phase P: x_proj = X * projW (K=64) -> sU cols 0..127 ----
  {
    v16h a0 = load_a_frag(&sX[(16 * w) * 64 + 0], 64, l);
    v16h a1 = load_a_frag(&sX[(16 * w) * 64 + 32], 64, l);
#pragma unroll
    for (int ct = 0; ct < 8; ++ct) {
      v8f c;
#pragma unroll
      for (int r = 0; r < 8; ++r) c[r] = 0.0f;
      c = wmma_f16(a0, load_b_frag(ProjT, 64, ct * 16, 0, l), c);
      c = wmma_f16(a1, load_b_frag(ProjT, 64, ct * 16, 32, l), c);
      int n = ct * 16 + lm;
#pragma unroll
      for (int r = 0; r < 8; ++r)
        sU[(16 * w + r + rowoff) * 256 + n] = (_Float16)c[r];
    }
  }
  // No barrier: each wave reads only its own row-tile of sU below.

  // ---- phase H2: H2 = relu(U * updW1 + ub1), K=256 ----
#pragma unroll
  for (int ct = 0; ct < 16; ++ct) {
    int n = ct * 16 + lm;
    float bb = ub1[n];
    v8f c;
#pragma unroll
    for (int r = 0; r < 8; ++r) c[r] = bb;
#pragma unroll
    for (int kk = 0; kk < 8; ++kk) {
      v16h a = load_a_frag(&sU[(16 * w) * 256 + kk * 32], 256, l);
      c = wmma_f16(a, load_b_frag(U1T, 256, ct * 16, kk * 32, l), c);
    }
#pragma unroll
    for (int r = 0; r < 8; ++r)
      sH2[(16 * w + r + rowoff) * 256 + n] = (_Float16)fmaxf(c[r], 0.0f);
  }

  // ---- phase OUT: out = elu(x_proj + H2 * updW2 + ub2) ----
#pragma unroll
  for (int ct = 0; ct < 8; ++ct) {
    int n = ct * 16 + lm;
    float bb = ub2[n];
    v8f c;
#pragma unroll
    for (int r = 0; r < 8; ++r) c[r] = bb;
#pragma unroll
    for (int kk = 0; kk < 8; ++kk) {
      v16h a = load_a_frag(&sH2[(16 * w) * 256 + kk * 32], 256, l);
      c = wmma_f16(a, load_b_frag(U2T, 256, ct * 16, kk * 32, l), c);
    }
#pragma unroll
    for (int r = 0; r < 8; ++r) {
      int m = 16 * w + r + rowoff;
      int g = n0 + m;
      if (g < Nn) {
        float xp = (float)sU[m * 256 + n];
        float v = xp + c[r];
        v = (v > 0.0f) ? v : expm1f(v);
        xout[(size_t)g * 128 + n] = v;
      }
    }
  }

  // ---- pos update ----
  if (t < 128) {
    int g = n0 + t;
    if (g < Nn) {
      float cc = cntv[g];
      cc = (cc < 1.0f) ? 1.0f : cc;
#pragma unroll
      for (int k = 0; k < 3; ++k)
        posout[(size_t)g * 3 + k] =
            pos[(size_t)g * 3 + k] + posSum[(size_t)g * 3 + k] / cc;
    }
  }
}

// ---------------------------------------------------------------------------
// Host launcher
// ---------------------------------------------------------------------------
static inline size_t align256(size_t v) { return (v + 255) & ~(size_t)255; }

extern "C" void kernel_launch(void* const* d_in, const int* in_sizes, int n_in,
                              void* d_out, int out_size, void* d_ws,
                              size_t ws_size, hipStream_t stream) {
  const float* x     = (const float*)d_in[0];
  const float* pos   = (const float*)d_in[1];
  const int*   eidx  = (const int*)d_in[2];
  const float* msgW1 = (const float*)d_in[3];
  const float* msgb1 = (const float*)d_in[4];
  const float* msgW2 = (const float*)d_in[5];
  const float* msgb2 = (const float*)d_in[6];
  const float* updW1 = (const float*)d_in[7];
  const float* updb1 = (const float*)d_in[8];
  const float* updW2 = (const float*)d_in[9];
  const float* updb2 = (const float*)d_in[10];
  const float* projW = (const float*)d_in[11];

  const int Nn = in_sizes[0] / 64;  // 50000
  const int E  = in_sizes[2] / 2;   // 1600000

  // workspace carve
  char* ws = (char*)d_ws;
  size_t o = 0;
  float* agg = (float*)(ws + o);      o += (size_t)Nn * 128 * sizeof(float);
  float* posSum = (float*)(ws + o);   o += (size_t)Nn * 3 * sizeof(float);
  float* cntv = (float*)(ws + o);     o += (size_t)Nn * sizeof(float);
  o = align256(o);
  _Float16* W1aT  = (_Float16*)(ws + o); o += (size_t)64 * 256 * 2;
  _Float16* W1bT  = (_Float16*)(ws + o); o += (size_t)64 * 256 * 2;
  _Float16* W2T   = (_Float16*)(ws + o); o += (size_t)256 * 128 * 2;
  _Float16* ProjT = (_Float16*)(ws + o); o += (size_t)64 * 128 * 2;
  _Float16* U1T   = (_Float16*)(ws + o); o += (size_t)256 * 256 * 2;
  _Float16* U2T   = (_Float16*)(ws + o); o += (size_t)256 * 128 * 2;
  o = align256(o);
  _Float16* xh    = (_Float16*)(ws + o); o += (size_t)Nn * 64 * 2;

  float* xout   = (float*)d_out;
  float* posout = xout + (size_t)Nn * 128;

  // zero accumulators (agg, posSum, cntv contiguous: Nn*132 floats)
  {
    size_t zn = (size_t)Nn * 132;
    zero_f32<<<(unsigned)((zn + 255) / 256), 256, 0, stream>>>(agg, zn);
  }
  // x -> f16 copy (gathered by both kernels; stays L2-resident)
  {
    size_t xn = (size_t)Nn * 64;
    f32_to_f16<<<(unsigned)((xn + 255) / 256), 256, 0, stream>>>(x, xh, xn);
  }
  // weight prep: transpose + f32->f16
  transpose_to_half<<<(64 * 256 + 255) / 256, 256, 0, stream>>>(msgW1, W1aT, 64, 256);
  transpose_to_half<<<(64 * 256 + 255) / 256, 256, 0, stream>>>(msgW1 + 64 * 256, W1bT, 64, 256);
  transpose_to_half<<<(256 * 128 + 255) / 256, 256, 0, stream>>>(msgW2, W2T, 256, 128);
  transpose_to_half<<<(64 * 128 + 255) / 256, 256, 0, stream>>>(projW, ProjT, 64, 128);
  transpose_to_half<<<(256 * 256 + 255) / 256, 256, 0, stream>>>(updW1, U1T, 256, 256);
  transpose_to_half<<<(256 * 128 + 255) / 256, 256, 0, stream>>>(updW2, U2T, 256, 128);

  // edge kernel
  {
    size_t smem = (size_t)(128 * 64) * 2 * 2    // sXi, sXj
                + (size_t)(128 * 256) * 2       // sH
                + (size_t)(128 + 384 + 128) * 4 // sDist, sPosIJ, sRowSq
                + (size_t)(256) * 4;            // sI, sJ
    egnn_edge_kernel<<<(E + 127) / 128, 256, smem, stream>>>(
        xh, pos, eidx, E, msgW1, msgb1, msgb2, W1aT, W1bT, W2T, agg, posSum,
        cntv);
  }
  // node kernel
  {
    size_t smem = (size_t)(128 * 64) * 2 + (size_t)(128 * 256) * 2 * 2;
    egnn_node_kernel<<<(Nn + 127) / 128, 256, smem, stream>>>(
        xh, pos, agg, posSum, cntv, updb1, updb2, ProjT, U1T, U2T, xout, posout,
        Nn);
  }
}
